// OnlineHadamard_50500225466612
// MI455X (gfx1250) — matrix operand order repaired and checked by reading, not compiled, not run on hardware
//
#include <hip/hip_runtime.h>
#include <math.h>

// bf16 / f32 ext-vector types matching the probe-confirmed WMMA builtin signature
typedef __attribute__((ext_vector_type(16))) __bf16 v16bf;
typedef __attribute__((ext_vector_type(8)))  __bf16 v8bf;
typedef __attribute__((ext_vector_type(2)))  __bf16 v2bf;
typedef __attribute__((ext_vector_type(8)))  float  v8f;

#define KD    172    // remainder Hadamard order
#define MD    64     // power-of-two FHT length
#define NROW  11008  // KD * MD
#define KPAD  192    // KD padded to 6 chunks of 32
#define JPAD  176    // KD padded to 11 tiles of 16
#define CSTR  200    // LDS row stride in bf16 elems (pad vs 192 to break bank conflicts)
#define XSTR  68     // LDS row stride in f32 for x staging (64 + 4 pad, keeps 16B align)

__global__ __launch_bounds__(256)
void hadamard_fused_kernel(const float* __restrict__ x,
                           const float* __restrict__ had,
                           float* __restrict__ out)
{
    __shared__ float  xs[KD * XSTR];        // 46,784 B: coalesced-staged x row-block
    __shared__ __bf16 hadB[JPAD * CSTR];    // 70,400 B: had, bf16, zero-padded 176x192
    __shared__ __bf16 Y1T[MD * CSTR];       // 25,600 B: FHT result, transposed [m][k]

    const int    t     = threadIdx.x;
    const size_t base  = (size_t)blockIdx.x * NROW;
    const float  scale = 1.0f / sqrtf((float)NROW);   // folds at compile time

    // ---- Stage A0: fully-coalesced copy of the 43KB x row-block into LDS ----
    // wave instruction = 32 lanes x 16B = 512B contiguous; full cacheline utilization
    {
        const float4* xin = (const float4*)(x + base);
        #pragma unroll
        for (int i = 0; i < 11; ++i) {
            const int idx = t + i * 256;            // float4 index within the row-block
            if (idx < NROW / 4) {
                float4 q = xin[idx];
                const int k = idx >> 4;             // row (64 floats = 16 float4 per row)
                const int m = (idx & 15) << 2;      // col
                *(float4*)&xs[k * XSTR + m] = q;    // ds_store_b128, 16B aligned
            }
        }
    }

    // ---- Stage A2: stage had -> LDS bf16, zero-padded to 176x192 (stride 200) ----
    for (int idx = t; idx < JPAD * (KPAD / 2); idx += 256) {
        const int row = idx / (KPAD / 2);
        const int kp  = (idx - row * (KPAD / 2)) * 2;     // even k
        float2 q = make_float2(0.0f, 0.0f);
        if (row < KD && kp < KD - 1)                      // pair (kp, kp+1) fully in range
            q = *(const float2*)(had + row * KD + kp);
        v2bf p;
        p[0] = (__bf16)q.x;
        p[1] = (__bf16)q.y;
        *(v2bf*)&hadB[row * CSTR + kp] = p;               // ds_store_b32
    }
    __syncthreads();

    // ---- Stage A1: per-row 64-pt FHT (exact fp32 butterflies), scale, bf16 -> Y1T ----
    if (t < KD) {
        float v[MD];
        #pragma unroll
        for (int i = 0; i < MD / 4; ++i) {                // conflict-free padded reads
            float4 q = *(const float4*)&xs[t * XSTR + 4 * i];   // ds_load_b128
            v[4*i+0] = q.x; v[4*i+1] = q.y; v[4*i+2] = q.z; v[4*i+3] = q.w;
        }
        #pragma unroll
        for (int h = 1; h < MD; h <<= 1) {
            #pragma unroll
            for (int i = 0; i < MD; ++i) {
                if ((i & h) == 0) {
                    float a = v[i], b = v[i + h];
                    v[i]     = a + b;
                    v[i + h] = a - b;
                }
            }
        }
        #pragma unroll
        for (int m = 0; m < MD; ++m)
            Y1T[m * CSTR + t] = (__bf16)(v[m] * scale);
    } else if (t < KPAD) {
        // zero-pad K rows 172..191 so the K=192 WMMA contraction is exact
        #pragma unroll
        for (int m = 0; m < MD; ++m)
            Y1T[m * CSTR + t] = (__bf16)0.0f;
    }
    __syncthreads();

    // ---- Stage B: Out = had(172x172) @ Y1(172x64), bf16 WMMA, f32 accum ----
    const int wave = t >> 5;
    const int lane = t & 31;
    const int half = lane >> 4;   // which 16-lane half of the wave
    const int lid  = lane & 15;
    float* orow = out + base;

    // 11 j-tiles of 16 rows (last partial), distributed over 8 waves
    for (int jt = wave; jt < 11; jt += 8) {
        v8f acc[4] = {};                       // 4 n-tiles of 16 cols
        const __bf16* arow = &hadB[(jt * 16 + lid) * CSTR];
        #pragma unroll
        for (int kc = 0; kc < 6; ++kc) {       // K = 192 in chunks of 32
            // A fragment (ISA 16-bit A 16x32 layout): lane row = jt*16+lid,
            //   elems 0..7  -> K = kc*32 + 8*half + {0..7}
            //   elems 8..15 -> K = kc*32 + 16 + 8*half + {0..7}
            const __bf16* ap = arow + kc * 32 + 8 * half;
            v8bf alo = *(const v8bf*)(ap);          // ds_load_b128 (16B aligned)
            v8bf ahi = *(const v8bf*)(ap + 16);     // ds_load_b128
            v16bf afrag = __builtin_shufflevector(alo, ahi,
                0, 1, 2, 3, 4, 5, 6, 7, 8, 9, 10, 11, 12, 13, 14, 15);
            #pragma unroll
            for (int nt = 0; nt < 4; ++nt) {
                // B fragment: lane col N = lid, elems 0..7 -> K = kc*32 + 8*half + {0..7}, elems 8..15 -> K = kc*32 + 16 + 8*half + {0..7}
                const __bf16* bp = &Y1T[(nt * 16 + lid) * CSTR + kc * 32 + 8 * half];
                v8bf blo = *(const v8bf*)(bp);      // ds_load_b128
                v8bf bhi = *(const v8bf*)(bp + 16); // ds_load_b128
                v16bf bfrag = __builtin_shufflevector(blo, bhi,
                    0, 1, 2, 3, 4, 5, 6, 7, 8, 9, 10, 11, 12, 13, 14, 15);
                acc[nt] = __builtin_amdgcn_wmma_f32_16x16x32_bf16(
                    /*neg_a=*/false, afrag, /*neg_b=*/false, bfrag,
                    /*c_mod=*/(short)0, acc[nt], /*reuse_a=*/false, /*reuse_b=*/false);
            }
        }
        // C/D layout: VGPR r, lane -> row = jt*16 + r + 8*half, col = nt*16 + lid
        if (jt != 10) {
            #pragma unroll
            for (int nt = 0; nt < 4; ++nt) {
                #pragma unroll
                for (int r = 0; r < 8; ++r) {
                    const int jj = jt * 16 + r + half * 8;
                    orow[(size_t)jj * MD + nt * 16 + lid] = acc[nt][r];
                }
            }
        } else {
            // partial tile: rows 160..175, keep only jj < 172
            #pragma unroll
            for (int nt = 0; nt < 4; ++nt) {
                #pragma unroll
                for (int r = 0; r < 8; ++r) {
                    const int jj = 160 + r + half * 8;
                    if (jj < KD)
                        orow[(size_t)jj * MD + nt * 16 + lid] = acc[nt][r];
                }
            }
        }
    }
}

extern "C" void kernel_launch(void* const* d_in, const int* in_sizes, int n_in,
                              void* d_out, int out_size, void* d_ws, size_t ws_size,
                              hipStream_t stream) {
    (void)n_in; (void)out_size; (void)d_ws; (void)ws_size;
    const float* x   = (const float*)d_in[0];   // [2,4096,11008] f32
    const float* had = (const float*)d_in[1];   // [172,172] f32
    float*       out = (float*)d_out;           // [2,4096,11008] f32

    const int rows = in_sizes[0] / NROW;        // 8192 row-blocks
    hadamard_fused_kernel<<<rows, 256, 0, stream>>>(x, had, out);
}
